// randomRNN_9397388443942
// MI455X (gfx1250) — matrix operand order repaired and checked by reading, not compiled, not run on hardware
//
#include <hip/hip_runtime.h>

// ---------------------------------------------------------------------------
// randomRNN for MI455X (gfx1250, wave32, WMMA f32<-f16 16x16x32)
//   T=256 B=1024 S=128 H=512 ; out = [T,B,2H] fp32
//   Part 1: persistent-kernel recurrent GEMM chain with device-wide barrier
//   Part 2: parallel NN2 GEMM over all T*B rows
// ---------------------------------------------------------------------------

#define T_DIM 256
#define B_DIM 1024
#define S_DIM 128
#define H_DIM 512
#define K_DIM 640 // S + H

#define F0 1.0f
#define F1 1.0f
#define F2 1.0f

typedef __attribute__((ext_vector_type(16))) _Float16 v16h;
typedef __attribute__((ext_vector_type(8)))  _Float16 h8v;
typedef __attribute__((ext_vector_type(8)))  float    v8f;
typedef __attribute__((ext_vector_type(4)))  float    f4v;

// gfx1250 has a hardware TANH trans op (co-executes with WMMA); use it if the
// toolchain declares the builtin, else fall back to ocml tanhf.
#if __has_builtin(__builtin_amdgcn_tanhf)
__device__ __forceinline__ float fast_tanh(float v) { return __builtin_amdgcn_tanhf(v); }
#else
__device__ __forceinline__ float fast_tanh(float v) { return tanhf(v); }
#endif

// ---- fragment loaders (layouts per CDNA5 ISA 7.12.2) ----------------------
// A (16-bit, 16x32): lane m=lane&15, g=lane>>4; VGPR p holds K = (p<4?0:16)+8g+2(p&3)+{0,1}
// => relative to rowp (already offset by k0 + 8*g), the fragment is two
//    contiguous 16-byte chunks: halves [0..7] and halves [16..23].

__device__ __forceinline__ v16h load_a_h(const _Float16* __restrict__ rowp) {
  h8v lo = *(const h8v*)(rowp);
  h8v hi = *(const h8v*)(rowp + 16);
  v16h a;
#pragma unroll
  for (int i = 0; i < 8; ++i) {
    a[i] = lo[i];
    a[8 + i] = hi[i];
  }
  return a;
}

__device__ __forceinline__ v16h load_a_f32_tanh(const float* __restrict__ rowp, float f) {
  f4v x0 = *(const f4v*)(rowp);
  f4v x1 = *(const f4v*)(rowp + 4);
  f4v x2 = *(const f4v*)(rowp + 16);
  f4v x3 = *(const f4v*)(rowp + 20);
  v16h a;
#pragma unroll
  for (int i = 0; i < 4; ++i) {
    a[i]      = (_Float16)fast_tanh(x0[i] * f);
    a[4 + i]  = (_Float16)fast_tanh(x1[i] * f);
    a[8 + i]  = (_Float16)fast_tanh(x2[i] * f);
    a[12 + i] = (_Float16)fast_tanh(x3[i] * f);
  }
  return a;
}

__device__ __forceinline__ v16h load_a_f32_scale(const float* __restrict__ rowp, float f) {
  f4v x0 = *(const f4v*)(rowp);
  f4v x1 = *(const f4v*)(rowp + 4);
  f4v x2 = *(const f4v*)(rowp + 16);
  f4v x3 = *(const f4v*)(rowp + 20);
  v16h a;
#pragma unroll
  for (int i = 0; i < 4; ++i) {
    a[i]      = (_Float16)(x0[i] * f);
    a[4 + i]  = (_Float16)(x1[i] * f);
    a[8 + i]  = (_Float16)(x2[i] * f);
    a[12 + i] = (_Float16)(x3[i] * f);
  }
  return a;
}

// B (16-bit, 32x16): lane n=lane&15, g=lane>>4; VGPR j holds K = (j<4?0:16)+8g+2(j&3)+{0,1}
// => relative to p (already offset to (row n, k0 + 8*g)): halves [0..7]
//    and [16..23] -> two ds_load_b128.
__device__ __forceinline__ v16h load_b_frag(const _Float16* __restrict__ p) {
  h8v lo = *(const h8v*)(p);
  h8v hi = *(const h8v*)(p + 16);
  v16h b;
#pragma unroll
  for (int i = 0; i < 8; ++i) {
    b[i] = lo[i];
    b[8 + i] = hi[i];
  }
  return b;
}

// ---- prep: zero h0, reset grid-barrier counter (every call; deterministic) --
__global__ void prep_kernel(_Float16* __restrict__ h0, unsigned* __restrict__ counter) {
  const size_t stride = (size_t)gridDim.x * blockDim.x;
  size_t i = (size_t)blockIdx.x * blockDim.x + threadIdx.x;
  unsigned* p = (unsigned*)h0;
  const size_t n = (size_t)B_DIM * H_DIM / 2; // f16 pairs
  for (size_t k = i; k < n; k += stride) p[k] = 0u;
  if (i == 0) *counter = 0u;
}

// ---- NN2: out[:, :, 512:1024] = tanh(x*f2 @ W2^T + b2) ---------------------
__global__ void __launch_bounds__(256) nn2_kernel(const float* __restrict__ x,
                                                  const float* __restrict__ W2,
                                                  const float* __restrict__ b2g,
                                                  float* __restrict__ out) {
  __shared__ _Float16 w2s[32 * 136]; // 32 rows of W2 tile, stride padded to 136
  const int tid = threadIdx.x;
  const int lane = tid & 31;
  const int wv = tid >> 5;
  const int n0 = (blockIdx.x >> 10) * 32;                // 16 column strips
  const int m0 = (((blockIdx.x & 1023) << 3) + wv) * 32; // 8192 row tiles

  for (int idx = tid; idx < 32 * 128; idx += 256) {
    const int r = idx >> 7, c = idx & 127;
    w2s[r * 136 + c] = (_Float16)W2[(n0 + r) * 128 + c];
  }
  __syncthreads();

  const int g = lane >> 4;
  const int l15 = lane & 15;

  v8f c00 = {}, c01 = {}, c10 = {}, c11 = {};
#pragma unroll
  for (int kc = 0; kc < 4; ++kc) {
    const int k0 = kc * 32;
    const float* r0 = x + (size_t)(m0 + l15) * S_DIM + k0 + 8 * g;
    v16h a0 = load_a_f32_scale(r0, F2);
    v16h a1 = load_a_f32_scale(r0 + 16 * S_DIM, F2);
    const _Float16* bp = &w2s[l15 * 136 + k0 + 8 * g];
    v16h fb0 = load_b_frag(bp);
    v16h fb1 = load_b_frag(bp + 16 * 136);
    c00 = __builtin_amdgcn_wmma_f32_16x16x32_f16(false, a0, false, fb0, (short)0, c00, false, false);
    c01 = __builtin_amdgcn_wmma_f32_16x16x32_f16(false, a0, false, fb1, (short)0, c01, false, false);
    c10 = __builtin_amdgcn_wmma_f32_16x16x32_f16(false, a1, false, fb0, (short)0, c10, false, false);
    c11 = __builtin_amdgcn_wmma_f32_16x16x32_f16(false, a1, false, fb1, (short)0, c11, false, false);
  }

  const float bias0 = b2g[n0 + l15];
  const float bias1 = b2g[n0 + 16 + l15];
#pragma unroll
  for (int r = 0; r < 8; ++r) {
    const size_t mA = (size_t)m0 + r + 8 * g;
    const size_t mB = mA + 16;
    const int nA = n0 + l15;
    const int nB = nA + 16;
    out[mA * 1024 + 512 + nA] = fast_tanh(c00[r] + bias0);
    out[mA * 1024 + 512 + nB] = fast_tanh(c01[r] + bias1);
    out[mB * 1024 + 512 + nA] = fast_tanh(c10[r] + bias0);
    out[mB * 1024 + 512 + nB] = fast_tanh(c11[r] + bias1);
  }
}

// ---- persistent recurrent kernel ------------------------------------------
// 64 blocks x 256 threads = 512 waves; each wave owns one 32x32 tile of h_new.
// Device-wide generation barrier between timesteps.
__global__ void __launch_bounds__(256) rnn_persistent(const float* __restrict__ x,
                                                      const float* __restrict__ W1,
                                                      const float* __restrict__ b1g,
                                                      float* __restrict__ out,
                                                      _Float16* __restrict__ hbuf0,
                                                      _Float16* __restrict__ hbuf1,
                                                      unsigned* __restrict__ counter) {
  __shared__ _Float16 w1s[32 * 648]; // 32 rows x 640 cols, padded stride 648 (~41.5 KB)
  const int tid = threadIdx.x;
  const int lane = tid & 31;
  const int wv = tid >> 5;
  const int n0 = (blockIdx.x >> 2) * 32;              // block-uniform column tile
  const int m0 = (((blockIdx.x & 3) << 3) + wv) * 32; // per-wave row tile

  // stage W1 tile once (reused for all 256 steps), f32 -> f16
  for (int idx = tid; idx < 32 * 640; idx += 256) {
    const int r = idx / 640, c = idx - r * 640;
    w1s[r * 648 + c] = (_Float16)W1[(n0 + r) * K_DIM + c];
  }
  __syncthreads();

  const int g = lane >> 4;
  const int l15 = lane & 15;
  const float bias0 = b1g[n0 + l15];
  const float bias1 = b1g[n0 + 16 + l15];
  const unsigned nblk = gridDim.x;

  for (int t = 0; t < T_DIM; ++t) {
    const _Float16* hp = (t & 1) ? hbuf1 : hbuf0;
    _Float16* hn = (t & 1) ? hbuf0 : hbuf1;
    const float* xt = x + (size_t)t * (B_DIM * S_DIM);

    v8f c00 = {}, c01 = {}, c10 = {}, c11 = {};

#pragma unroll
    for (int kc = 0; kc < 20; ++kc) {
      const int k0 = kc * 32;
      v16h a0, a1;
      if (k0 < S_DIM) { // A columns from tanh(x_t * f0), converted on the fly
        const float* r0 = xt + (size_t)(m0 + l15) * S_DIM + k0 + 8 * g;
        a0 = load_a_f32_tanh(r0, F0);
        a1 = load_a_f32_tanh(r0 + 16 * S_DIM, F0);
      } else { // A columns from h (f16 double buffer)
        const _Float16* r0 = hp + (size_t)(m0 + l15) * H_DIM + (k0 - S_DIM) + 8 * g;
        a0 = load_a_h(r0);
        a1 = load_a_h(r0 + 16 * H_DIM);
      }
      const _Float16* bp = &w1s[l15 * 648 + k0 + 8 * g];
      v16h fb0 = load_b_frag(bp);
      v16h fb1 = load_b_frag(bp + 16 * 648);
      c00 = __builtin_amdgcn_wmma_f32_16x16x32_f16(false, a0, false, fb0, (short)0, c00, false, false);
      c01 = __builtin_amdgcn_wmma_f32_16x16x32_f16(false, a0, false, fb1, (short)0, c01, false, false);
      c10 = __builtin_amdgcn_wmma_f32_16x16x32_f16(false, a1, false, fb0, (short)0, c10, false, false);
      c11 = __builtin_amdgcn_wmma_f32_16x16x32_f16(false, a1, false, fb1, (short)0, c11, false, false);
    }

    // epilogue: bias + tanh, write fp32 output half and f16 h_next
    float* outt = out + (size_t)t * ((size_t)B_DIM * 2 * H_DIM);
#pragma unroll
    for (int r = 0; r < 8; ++r) {
      const size_t mA = (size_t)m0 + r + 8 * g;
      const size_t mB = mA + 16;
      const int nA = n0 + l15;
      const int nB = nA + 16;
      const float vAA = fast_tanh(c00[r] + bias0);
      const float vAB = fast_tanh(c01[r] + bias1);
      const float vBA = fast_tanh(c10[r] + bias0);
      const float vBB = fast_tanh(c11[r] + bias1);
      outt[mA * (2 * H_DIM) + nA] = vAA;
      outt[mA * (2 * H_DIM) + nB] = vAB;
      outt[mB * (2 * H_DIM) + nA] = vBA;
      outt[mB * (2 * H_DIM) + nB] = vBB;
      hn[mA * H_DIM + nA] = (_Float16)(vAA * F1);
      hn[mA * H_DIM + nB] = (_Float16)(vAB * F1);
      hn[mB * H_DIM + nA] = (_Float16)(vBA * F1);
      hn[mB * H_DIM + nB] = (_Float16)(vBB * F1);
    }

    // ---- device-wide barrier: monotonic counter, target = (t+1)*nblk ----
    __threadfence(); // make h_next visible at agent scope
    __syncthreads();
    if (tid == 0) {
      __hip_atomic_fetch_add(counter, 1u, __ATOMIC_ACQ_REL, __HIP_MEMORY_SCOPE_AGENT);
      const unsigned target = (unsigned)(t + 1) * nblk;
      while (__hip_atomic_load(counter, __ATOMIC_ACQUIRE, __HIP_MEMORY_SCOPE_AGENT) < target) {
        __builtin_amdgcn_s_sleep(8);
      }
    }
    __syncthreads();
    __threadfence(); // acquire: don't read stale h through local caches
  }
}

// ---------------------------------------------------------------------------
extern "C" void kernel_launch(void* const* d_in, const int* in_sizes, int n_in,
                              void* d_out, int out_size, void* d_ws, size_t ws_size,
                              hipStream_t stream) {
  (void)in_sizes; (void)n_in; (void)out_size; (void)ws_size;
  const float* x  = (const float*)d_in[0];
  const float* W1 = (const float*)d_in[1];
  const float* b1 = (const float*)d_in[2];
  const float* W2 = (const float*)d_in[3];
  const float* b2 = (const float*)d_in[4];
  float* out = (float*)d_out;

  // workspace layout: [counter (64B)] [h buf0 1MiB f16] [h buf1 1MiB f16]
  unsigned* counter = (unsigned*)d_ws;
  _Float16* h0 = (_Float16*)((char*)d_ws + 64);
  _Float16* h1 = h0 + (size_t)B_DIM * H_DIM;

  prep_kernel<<<256, 256, 0, stream>>>(h0, counter);

  // NN2: (T*B/32)/8 row-blocks per column strip * 16 strips = 16384 blocks
  nn2_kernel<<<16384, 256, 0, stream>>>(x, W2, b2, out);

  // persistent recurrent chain: 64 blocks (one 32x32 tile per wave)
  rnn_persistent<<<64, 256, 0, stream>>>(x, W1, b1, out, h0, h1, counter);
}
